// SpatialAttention_14534169330039
// MI455X (gfx1250) — matrix operand order repaired and checked by reading, not compiled, not run on hardware
//
#include <hip/hip_runtime.h>
#include <hip/hip_bf16.h>

typedef __attribute__((ext_vector_type(16))) __bf16 v16bf;
typedef __attribute__((ext_vector_type(8)))  float  v8f;

#define N_TOK  4096
#define C_IN   256
#define C_OUT  128

union BF16x16 { uint4 u[2]; v16bf v; };

// 16 contiguous bf16 (two b128 loads) -> B-matrix fragment (lane n=l&15, K=kbase..kbase+15)
__device__ inline v16bf ld_bf16x16(const __bf16* p) {
  BF16x16 t;
  t.u[0] = *reinterpret_cast<const uint4*>(p);
  t.u[1] = *reinterpret_cast<const uint4*>(p + 8);
  return t.v;
}

// A-matrix fragment: elements 0..7 = K at +0..+7, elements 8..15 = K at +16..+23 (base pre-offset by koffA)
__device__ inline v16bf ld_frag_split(const __bf16* p) {
  BF16x16 t;
  t.u[0] = *reinterpret_cast<const uint4*>(p);
  t.u[1] = *reinterpret_cast<const uint4*>(p + 16);
  return t.v;
}

// ---------------------------------------------------------------------------
// Stage 1: q = Wq * p + bq, k = Wk * b + bk, both stored bf16 as [B, N, 128]
//   out[n,o] = sum_c X[c,n] * W[o,c] + bias[o]
//   WMMA: A = W tile [16 o x 32 c] (contiguous rows), B = X tile [32 c x 16 n]
// ---------------------------------------------------------------------------
__global__ __launch_bounds__(128) void proj_kernel(
    const float* __restrict__ p, const float* __restrict__ bsrc,
    const float* __restrict__ Wq, const float* __restrict__ bq,
    const float* __restrict__ Wk, const float* __restrict__ bk,
    __bf16* __restrict__ q_ws, __bf16* __restrict__ k_ws) {
  const int lane = threadIdx.x & 31;
  const int wave = threadIdx.x >> 5;
  const int tile = blockIdx.x * 4 + wave;
  const bool is_k = blockIdx.y != 0;
  const float* X    = is_k ? bsrc : p;
  const float* W    = is_k ? Wk   : Wq;
  const float* bias = is_k ? bk   : bq;
  __bf16* outp      = is_k ? k_ws : q_ws;

  const int o_t = tile & 7;           // 128/16 o-tiles
  const int n_t = (tile >> 3) & 255;  // 4096/16 n-tiles
  const int bat = tile >> 11;         // 4 batches

  const float* Xb = X + (size_t)bat * C_IN * N_TOK;
  __bf16* outb = outp + (size_t)bat * N_TOK * C_OUT;

  const int lm    = lane & 15;
  const int hi    = lane >> 4;
  const int o_row = o_t * 16 + lm;    // A-matrix row (M = l&15 for all lanes)
  const int n_col = n_t * 16 + lm;    // B-matrix column
  const int koffA = hi ? 8 : 0;       // A K offsets
  const int kbB   = hi ? 16 : 0;      // B K base

  v8f acc = {};
  #pragma unroll
  for (int s = 0; s < 8; ++s) {       // K = C_IN = 256, 8 steps of 32
    const float* wrow = W + (size_t)o_row * C_IN + s * 32 + koffA;
    float4 f0 = *reinterpret_cast<const float4*>(wrow);
    float4 f1 = *reinterpret_cast<const float4*>(wrow + 4);
    float4 f2 = *reinterpret_cast<const float4*>(wrow + 16);
    float4 f3 = *reinterpret_cast<const float4*>(wrow + 20);
    v16bf a;
    a[0]=(__bf16)f0.x;  a[1]=(__bf16)f0.y;  a[2]=(__bf16)f0.z;  a[3]=(__bf16)f0.w;
    a[4]=(__bf16)f1.x;  a[5]=(__bf16)f1.y;  a[6]=(__bf16)f1.z;  a[7]=(__bf16)f1.w;
    a[8]=(__bf16)f2.x;  a[9]=(__bf16)f2.y;  a[10]=(__bf16)f2.z; a[11]=(__bf16)f2.w;
    a[12]=(__bf16)f3.x; a[13]=(__bf16)f3.y; a[14]=(__bf16)f3.z; a[15]=(__bf16)f3.w;
    v16bf bb;
    #pragma unroll
    for (int e = 0; e < 16; ++e)      // gather column n_col over channels
      bb[e] = (__bf16)Xb[(size_t)(s * 32 + koffA + (e & 7) + ((e >> 3) << 4)) * N_TOK + n_col];
    acc = __builtin_amdgcn_wmma_f32_16x16x32_bf16(false, a, false, bb,
                                                  (short)0, acc, false, false);
  }
  // D layout: lane column n = lm; rows o = o_t*16 + j + hi*8. Store transposed -> [n, o]
  #pragma unroll
  for (int j = 0; j < 8; ++j) {
    int o = o_t * 16 + j + hi * 8;
    float v = acc[j] + bias[o];
    outb[(size_t)n_col * C_OUT + o] = (__bf16)v;
  }
}

// ---------------------------------------------------------------------------
// Stage 1b: V = bf16(b) kept in [B, C, N] layout (keys contiguous per channel)
// ---------------------------------------------------------------------------
__global__ void cast_v_kernel(const float* __restrict__ src,
                              __bf16* __restrict__ dst, int n4) {
  int i = blockIdx.x * blockDim.x + threadIdx.x;
  if (i >= n4) return;
  float4 f = reinterpret_cast<const float4*>(src)[i];
  union { __bf16 h[4]; unsigned long long u; } t;
  t.h[0]=(__bf16)f.x; t.h[1]=(__bf16)f.y; t.h[2]=(__bf16)f.z; t.h[3]=(__bf16)f.w;
  reinterpret_cast<unsigned long long*>(dst)[i] = t.u;
}

// ---------------------------------------------------------------------------
// Stage 2: fused flash attention. One wave = 16 queries, 4 waves/WG (64 q/WG).
// Per 32-key chunk: 8 energy WMMA + online softmax + 16 PV WMMA (C=256).
// ---------------------------------------------------------------------------
__global__ __launch_bounds__(128) void flash_kernel(
    const __bf16* __restrict__ qg, const __bf16* __restrict__ kg,
    const __bf16* __restrict__ vg, float* __restrict__ outg) {
  __shared__ __align__(16) __bf16 lds_p[4][16][32];

  const int lane = threadIdx.x & 31;
  const int wave = threadIdx.x >> 5;
  const int bat  = blockIdx.x >> 6;   // 4 batches
  const int qt   = blockIdx.x & 63;   // 64 query tiles of 64
  const int q0   = qt * 64 + wave * 16;

  const __bf16* qb = qg + (size_t)bat * N_TOK * C_OUT;
  const __bf16* kb = kg + (size_t)bat * N_TOK * C_OUT;
  const __bf16* vb = vg + (size_t)bat * C_IN  * N_TOK;
  float* outb = outg + (size_t)bat * C_IN * N_TOK;

  const int lm    = lane & 15;
  const int hi    = lane >> 4;
  const int koffA = hi ? 8 : 0;
  const int kbB   = hi ? 16 : 0;

  // Preload Q fragments: 16 rows x K=128 (4 x K32 fragments)
  v16bf qf[4];
  #pragma unroll
  for (int s = 0; s < 4; ++s)
    qf[s] = ld_frag_split(qb + (size_t)(q0 + lm) * C_OUT + s * 32 + koffA);

  v8f zero = {};
  v8f acc[16];
  #pragma unroll
  for (int ct = 0; ct < 16; ++ct) acc[ct] = zero;
  float m8[8], l8[8];
  #pragma unroll
  for (int j = 0; j < 8; ++j) { m8[j] = -1e30f; l8[j] = 0.0f; }

  for (int key0 = 0; key0 < N_TOK; key0 += 32) {
    if (key0 + 32 < N_TOK) {
      __builtin_prefetch(kb + (size_t)(key0 + 32 + lm) * C_OUT, 0, 3);
    }
    // energy tile 16 x 32 (two 16x16 halves), K-dim = 128
    v8f e0 = zero, e1 = zero;
    #pragma unroll
    for (int s = 0; s < 4; ++s) {
      v16bf b0 = ld_frag_split(kb + (size_t)(key0 + lm) * C_OUT + s * 32 + koffA);
      e0 = __builtin_amdgcn_wmma_f32_16x16x32_bf16(false, qf[s], false, b0,
                                                   (short)0, e0, false, false);
      v16bf b1 = ld_frag_split(kb + (size_t)(key0 + 16 + lm) * C_OUT + s * 32 + koffA);
      e1 = __builtin_amdgcn_wmma_f32_16x16x32_bf16(false, qf[s], false, b1,
                                                   (short)0, e1, false, false);
    }
    // online softmax: C/D layout -> row j (+hi*8) spread over the 16-lane group
    float alpha[8];
    #pragma unroll
    for (int j = 0; j < 8; ++j) {
      float cm = fmaxf(e0[j], e1[j]);
      #pragma unroll
      for (int msk = 1; msk < 16; msk <<= 1)
        cm = fmaxf(cm, __shfl_xor(cm, msk, 32));
      float nm = fmaxf(m8[j], cm);
      alpha[j] = __expf(m8[j] - nm);
      m8[j] = nm;
      float p0 = __expf(e0[j] - nm);
      float p1 = __expf(e1[j] - nm);
      e0[j] = p0; e1[j] = p1;
      float cs = p0 + p1;
      #pragma unroll
      for (int msk = 1; msk < 16; msk <<= 1)
        cs += __shfl_xor(cs, msk, 32);
      l8[j] = l8[j] * alpha[j] + cs;
    }
    // transpose P from C/D layout to A layout via wave-private LDS
    #pragma unroll
    for (int j = 0; j < 8; ++j) {
      lds_p[wave][hi * 8 + j][lm]      = (__bf16)e0[j];
      lds_p[wave][hi * 8 + j][lm + 16] = (__bf16)e1[j];
    }
    __asm__ volatile("s_wait_dscnt 0" ::: "memory");
    v16bf af = ld_frag_split(&lds_p[wave][lm][koffA]);
    // rescale accumulators by alpha (per row)
    #pragma unroll
    for (int ct = 0; ct < 16; ++ct) {
      #pragma unroll
      for (int j = 0; j < 8; ++j) acc[ct][j] *= alpha[j];
    }
    // PV: attn [16 x 32] x V [32 keys x 256 C] (keys contiguous in V[B,C,N])
    #pragma unroll
    for (int ct = 0; ct < 16; ++ct) {
      v16bf bv = ld_frag_split(vb + (size_t)(ct * 16 + lm) * N_TOK + key0 + koffA);
      acc[ct] = __builtin_amdgcn_wmma_f32_16x16x32_bf16(false, af, false, bv,
                                                        (short)0, acc[ct], false, false);
    }
  }
  // epilogue: out[b, c, n] = acc / l   (lane col = c, row j(+hi*8) = query n)
  #pragma unroll
  for (int ct = 0; ct < 16; ++ct) {
    int c = ct * 16 + lm;
    #pragma unroll
    for (int j = 0; j < 8; ++j) {
      int n = q0 + hi * 8 + j;
      outb[(size_t)c * N_TOK + n] = acc[ct][j] * (1.0f / l8[j]);
    }
  }
}

extern "C" void kernel_launch(void* const* d_in, const int* in_sizes, int n_in,
                              void* d_out, int out_size, void* d_ws, size_t ws_size,
                              hipStream_t stream) {
  (void)in_sizes; (void)n_in; (void)out_size; (void)ws_size;
  const float* p  = (const float*)d_in[0];
  const float* b  = (const float*)d_in[1];
  const float* Wq = (const float*)d_in[2];
  const float* bq = (const float*)d_in[3];
  const float* Wk = (const float*)d_in[4];
  const float* bk = (const float*)d_in[5];
  float* out = (float*)d_out;

  char* ws = (char*)d_ws;
  __bf16* q_ws = (__bf16*)(ws);                              // 4 MB: [4,4096,128] bf16
  __bf16* k_ws = (__bf16*)(ws + (size_t)4  * 1024 * 1024);   // 4 MB: [4,4096,128] bf16
  __bf16* v_ws = (__bf16*)(ws + (size_t)8  * 1024 * 1024);   // 8 MB: [4,256,4096] bf16

  // q & k projections: 4*256*8 = 8192 tiles per tensor, 4 waves per block
  dim3 pg(2048, 2, 1);
  proj_kernel<<<pg, 128, 0, stream>>>(p, b, Wq, bq, Wk, bk, q_ws, k_ws);

  // V cast: 4*256*4096 = 4194304 floats = 1048576 float4
  cast_v_kernel<<<4096, 256, 0, stream>>>(b, v_ws, 1048576);

  // fused attention: 4 batches * 64 query tiles
  flash_kernel<<<256, 128, 0, stream>>>(q_ws, k_ws, v_ws, out);
}
